// Basic097_24086176596247
// MI455X (gfx1250) — matrix operand order repaired and checked by reading, not compiled, not run on hardware
//
#include <hip/hip_runtime.h>
#include <hip/hip_bf16.h>
#include <hip/hip_fp16.h>

#define VOCAB 32000
#define EMB   128
#define HID   512
#define BATCH 64
#define SEQ   1024

#define NWG        16          // persistent workgroups
#define COLS_PER_WG 32         // HID / NWG
#define NT_PER_WG   2          // 16-col tiles per WG
#define KT_TOTAL    20         // (EMB+HID)/32
#define KT_X        4          // EMB/32
#define KT_H        16         // HID/32

typedef __attribute__((ext_vector_type(16))) _Float16 v16h;
typedef __attribute__((ext_vector_type(8)))  _Float16 v8h;
typedef __attribute__((ext_vector_type(8)))  float    v8f;

// ---------------- workspace layout (bytes) ----------------
#define WS_X16    0                      // packed A tiles [SEQ][4mt*4kt*512] f16 : 16777216
#define WS_WPACK  16777216               // 4*32*20*512 f16     : 2621440
#define WS_U      19398656               // 513 f32 (u + c0)    : pad 4096
#define WS_HBUF   19402752               // 2* packed H (64KB)  : 131072
#define WS_YPART  19533824               // [SEQ,BATCH,NWG] f32 : 4194304
#define WS_BAR    23728128               // int

// Packed A-operand index (halfs) for a 16-bit 16x32 WMMA A tile set.
// Per tile (512 halfs): [0..255] = per-lane halfs i0..7 (lane*8), [256..511] = i8..15.
// Element (M,K): mt=M/16, kt=K/32, kl=K%32, grp=kl/8:
//   lane = (M&15) + ((grp&1)<<4), block = grp>>1, pos = kl&7
__device__ __host__ __forceinline__ int packA_idx(int M, int K, int ktiles) {
  int mt = M >> 4, kt = K >> 5, kl = K & 31, grp = kl >> 3;
  int lane = (M & 15) + ((grp & 1) << 4);
  return ((mt * ktiles + kt) * 512) + ((grp >> 1) << 8) + lane * 8 + (kl & 7);
}

// ---------------- pre-pass kernels ----------------

// x packed: x16[(s*B+b)*E + packA_idx(b, e, 4)] = (f16) emb[inputs[b][s]][e]
__global__ void k_embed(const int* __restrict__ inp, const float* __restrict__ emb,
                        _Float16* __restrict__ x16) {
  int sb = blockIdx.x;                 // s*BATCH + b
  int s = sb / BATCH, b = sb % BATCH;
  int e = threadIdx.x;
  int tok = inp[b * SEQ + s];
  x16[(size_t)sb * EMB + packA_idx(b, e, KT_X)] =
      (_Float16)emb[(size_t)tok * EMB + e];
}

// B-tile pack: 16-bit B 32x16 tile: lane l, half i: K_local = (i<8 ? i : 16+i-8) + (l>=16?8:0), N = l&15
__global__ void k_pack(const float* __restrict__ Wx0, const float* __restrict__ Wh0,
                       const float* __restrict__ Wx1, const float* __restrict__ Wh1,
                       const float* __restrict__ Wx2, const float* __restrict__ Wh2,
                       const float* __restrict__ Wx3, const float* __restrict__ Wh3,
                       _Float16* __restrict__ wpack) {
  int tile = blockIdx.x;               // g*(32*20) + nt*20 + kt
  int g   = tile / (32 * KT_TOTAL);
  int rem = tile % (32 * KT_TOTAL);
  int nt = rem / KT_TOTAL, kt = rem % KT_TOTAL;
  int l = threadIdx.x >> 4, i = threadIdx.x & 15;
  int K = kt * 32 + ((i & 8) << 1) + ((l & 16) >> 1) + (i & 7);
  int n = nt * 16 + (l & 15);
  const float* Wx = (g == 0) ? Wx0 : (g == 1) ? Wx1 : (g == 2) ? Wx2 : Wx3;
  const float* Wh = (g == 0) ? Wh0 : (g == 1) ? Wh1 : (g == 2) ? Wh2 : Wh3;
  float v = (K < EMB) ? Wx[(size_t)K * HID + n] : Wh[(size_t)(K - EMB) * HID + n];
  wpack[(size_t)tile * 512 + threadIdx.x] = (_Float16)v;
}

// u[h] = W_hq[h,:].dense_w ; u[HID] = c0 = b_q.dense_w + dense_b
__global__ void k_u(const float* __restrict__ W_hq, const float* __restrict__ b_q,
                    const float* __restrict__ dw, const float* __restrict__ db,
                    float* __restrict__ u) {
  int h = threadIdx.x;                 // 512 threads
  float a = 0.f;
  for (int e = 0; e < EMB; ++e) a += W_hq[(size_t)h * EMB + e] * dw[e];
  u[h] = a;
  if (h == 0) {
    float c = db[0];
    for (int e = 0; e < EMB; ++e) c += b_q[e] * dw[e];
    u[HID] = c;
  }
}

// H0 -> packed f16 hbuf[0]; zero barrier counter every launch (determinism)
__global__ void k_init(const float* __restrict__ H0, _Float16* __restrict__ hbuf,
                       int* __restrict__ bar) {
  int t = blockIdx.x * blockDim.x + threadIdx.x;
  if (t < BATCH * HID) {
    int b = t / HID, n = t % HID;
    hbuf[packA_idx(b, n, KT_H)] = (_Float16)H0[t];
  }
  if (t == 0) *bar = 0;
}

// ---------------- persistent recurrent kernel ----------------

__device__ __forceinline__ void async_wait0() {
#if __has_builtin(__builtin_amdgcn_s_wait_asynccnt)
  __builtin_amdgcn_s_wait_asynccnt(0);
#else
  asm volatile("s_wait_asynccnt 0" ::: "memory");
#endif
}

__global__ void __launch_bounds__(256, 1)
k_lstm(const _Float16* __restrict__ x16, const _Float16* __restrict__ wpack,
       const float* __restrict__ u, _Float16* __restrict__ hbuf,
       float* __restrict__ ypart, int* __restrict__ bar,
       const float* __restrict__ C0,
       const float* __restrict__ b_i, const float* __restrict__ b_f,
       const float* __restrict__ b_o, const float* __restrict__ b_c,
       float* __restrict__ out) {
  // 64KB: packed H A-tiles overlaid with gate/yred scratch (phases separated by barriers)
  __shared__ union SM {
    _Float16 h[4 * KT_H * 512];                                   // 65536 B
    struct { float gates[4][BATCH][COLS_PER_WG]; float yred[BATCH][4]; } g;
  } sm;

  const int wg   = blockIdx.x;
  const int tid  = threadIdx.x;
  const int wave = tid >> 5;
  const int lane = tid & 31;
  const int g    = wave >> 1;          // gate: 0=i 1=f 2=o 3=c
  const int ln   = wave & 1;           // local 16-col tile
  const int nt   = wg * NT_PER_WG + ln;
  const int ml   = lane & 15;

  // LDS byte offset of sm (generic addr low 32 bits == LDS address, ISA §10.2)
  const uint32_t smbase = (uint32_t)(uintptr_t)(void*)&sm;

  // pin this wave's weight slice (20 K-tiles) in VGPRs for all 1024 steps
  v16h wB[KT_TOTAL];
  {
    const _Float16* wp = wpack + (size_t)((g * 32 + nt) * KT_TOTAL) * 512 + lane * 16;
#pragma unroll
    for (int kt = 0; kt < KT_TOTAL; ++kt)
      wB[kt] = *(const v16h*)(wp + (size_t)kt * 512);
  }
  const float* bp = (g == 0) ? b_i : (g == 1) ? b_f : (g == 2) ? b_o : b_c;
  const float bias = bp[nt * 16 + ml];

  // elementwise ownership: thread -> (batch row b, 8 of the WG's 32 columns)
  const int b = tid >> 2, q = tid & 3;
  float Creg[8], ureg[8];
#pragma unroll
  for (int j = 0; j < 8; ++j) {
    int n = wg * COLS_PER_WG + q * 8 + j;
    Creg[j] = C0[(size_t)b * HID + n];
    ureg[j] = u[n];
  }
  // packed-H store offset for this thread's 8 columns (one contiguous v8h)
  const int hoff = ((b >> 4) * KT_H + wg) * 512 + ((q >> 1) << 8) +
                   ((b & 15) + ((q & 1) << 4)) * 8;

  for (int s = 0; s < SEQ; ++s) {
    const _Float16* Hcur = hbuf + (size_t)(s & 1) * BATCH * HID;
    _Float16*       Hnxt = hbuf + (size_t)((s + 1) & 1) * BATCH * HID;
    const _Float16* xb   = x16 + (size_t)s * BATCH * EMB;

    // prefetch next step's packed x block into cache
    if (s + 1 < SEQ)
      __builtin_prefetch(x16 + (size_t)(s + 1) * BATCH * EMB + tid * 32, 0, 0);

    // async copy of packed H (64KB) straight into LDS: 256B per thread, 16x b128
    {
      uint32_t la = smbase + (uint32_t)tid * 256u;
      uint64_t ga = (uint64_t)(uintptr_t)((const char*)Hcur + tid * 256);
#pragma unroll
      for (int c = 0; c < 16; ++c) {
        asm volatile("global_load_async_to_lds_b128 %0, %1, off"
                     :: "v"(la + c * 16u), "v"(ga + (uint64_t)(c * 16))
                     : "memory");
      }
    }

    v8f acc[4];
#pragma unroll
    for (int mt = 0; mt < 4; ++mt)
#pragma unroll
      for (int r = 0; r < 8; ++r) acc[mt][r] = 0.f;

    // K segment 1: x_t (K=0..127): packed A tiles from global (overlaps async copy)
#pragma unroll
    for (int kt = 0; kt < KT_X; ++kt) {
#pragma unroll
      for (int mt = 0; mt < 4; ++mt) {
        const _Float16* tb = xb + (mt * KT_X + kt) * 512;
        v8h lo = *(const v8h*)(tb + lane * 8);
        v8h hi = *(const v8h*)(tb + 256 + lane * 8);
        v16h a = __builtin_shufflevector(lo, hi, 0,1,2,3,4,5,6,7,8,9,10,11,12,13,14,15);
        acc[mt] = __builtin_amdgcn_wmma_f32_16x16x32_f16(
            false, a, false, wB[kt], (short)0, acc[mt], false, false);
      }
    }

    async_wait0();
    __syncthreads();   // packed H now fully staged in LDS

    // K segment 2: H_{t-1} (K=128..639): A tiles from LDS, bank-conflict-free
#pragma unroll
    for (int kt = 0; kt < KT_H; ++kt) {
#pragma unroll
      for (int mt = 0; mt < 4; ++mt) {
        const _Float16* tb = sm.h + (mt * KT_H + kt) * 512;
        v8h lo = *(const v8h*)(tb + lane * 8);
        v8h hi = *(const v8h*)(tb + 256 + lane * 8);
        v16h a = __builtin_shufflevector(lo, hi, 0,1,2,3,4,5,6,7,8,9,10,11,12,13,14,15);
        acc[mt] = __builtin_amdgcn_wmma_f32_16x16x32_f16(
            false, a, false, wB[KT_X + kt], (short)0, acc[mt], false, false);
      }
    }
    __syncthreads();   // all waves done reading sm.h -> safe to overlay gates

    // stage gate pre-activations (C/D layout: M = mt*16 + r + (lane>=16?8:0), N = ml)
#pragma unroll
    for (int mt = 0; mt < 4; ++mt)
#pragma unroll
      for (int r = 0; r < 8; ++r) {
        int m = mt * 16 + r + ((lane & 16) ? 8 : 0);
        sm.g.gates[g][m][ln * 16 + ml] = acc[mt][r] + bias;
      }
    __syncthreads();

    // elementwise gate math + C/H update + partial output dot
    float yp = 0.f;
    v8h hv;
#pragma unroll
    for (int j = 0; j < 8; ++j) {
      int nl = q * 8 + j;
      float gi = sm.g.gates[0][b][nl], gf = sm.g.gates[1][b][nl];
      float go = sm.g.gates[2][b][nl], gc = sm.g.gates[3][b][nl];
      float I  = 1.f / (1.f + __expf(-gi));
      float F  = 1.f / (1.f + __expf(-gf));
      float O  = 1.f / (1.f + __expf(-go));
      float Ct = tanhf(gc);
      float Cn = F * Creg[j] + I * Ct;
      float Hn = O * tanhf(Cn);
      Creg[j] = Cn;
      hv[j] = (_Float16)Hn;
      yp += Hn * ureg[j];
      if (s == SEQ - 1) {
        int n = wg * COLS_PER_WG + nl;
        out[(size_t)SEQ * BATCH + (size_t)b * HID + n] = Hn;                 // H
        out[(size_t)SEQ * BATCH + BATCH * HID + (size_t)b * HID + n] = Cn;   // C
      }
    }
    *(v8h*)(Hnxt + hoff) = hv;          // one contiguous 16B packed store
    sm.g.yred[b][q] = yp;
    __syncthreads();
    if (tid < BATCH)
      ypart[((size_t)s * BATCH + tid) * NWG + wg] =
          sm.g.yred[tid][0] + sm.g.yred[tid][1] + sm.g.yred[tid][2] + sm.g.yred[tid][3];

    // device-wide barrier (16 co-resident WGs)
    __threadfence();
    __syncthreads();
    if (tid == 0) {
      __hip_atomic_fetch_add(bar, 1, __ATOMIC_RELEASE, __HIP_MEMORY_SCOPE_AGENT);
      int target = NWG * (s + 1);
      while (__hip_atomic_load(bar, __ATOMIC_ACQUIRE, __HIP_MEMORY_SCOPE_AGENT) < target)
        __builtin_amdgcn_s_sleep(1);
    }
    __syncthreads();
    __threadfence();
  }
}

// out[s*B+b] = c0 + sum_wg ypart
__global__ void k_reduce(const float* __restrict__ ypart, const float* __restrict__ u,
                         float* __restrict__ out) {
  int i = blockIdx.x * blockDim.x + threadIdx.x;   // < SEQ*BATCH
  float a = u[HID];
#pragma unroll
  for (int w = 0; w < NWG; ++w) a += ypart[(size_t)i * NWG + w];
  out[i] = a;
}

// ---------------- launcher ----------------

extern "C" void kernel_launch(void* const* d_in, const int* in_sizes, int n_in,
                              void* d_out, int out_size, void* d_ws, size_t ws_size,
                              hipStream_t stream) {
  const int*   inputs = (const int*)  d_in[0];
  const float* H0     = (const float*)d_in[1];
  const float* C0     = (const float*)d_in[2];
  const float* emb    = (const float*)d_in[3];
  const float* W_xi = (const float*)d_in[4],  *W_hi = (const float*)d_in[5],  *b_i = (const float*)d_in[6];
  const float* W_xf = (const float*)d_in[7],  *W_hf = (const float*)d_in[8],  *b_f = (const float*)d_in[9];
  const float* W_xo = (const float*)d_in[10], *W_ho = (const float*)d_in[11], *b_o = (const float*)d_in[12];
  const float* W_xc = (const float*)d_in[13], *W_hc = (const float*)d_in[14], *b_c = (const float*)d_in[15];
  const float* W_hq = (const float*)d_in[16], *b_q  = (const float*)d_in[17];
  const float* dw   = (const float*)d_in[18], *db   = (const float*)d_in[19];
  float* out = (float*)d_out;

  char* ws = (char*)d_ws;
  _Float16* x16   = (_Float16*)(ws + WS_X16);
  _Float16* wpack = (_Float16*)(ws + WS_WPACK);
  float*    u     = (float*)   (ws + WS_U);
  _Float16* hbuf  = (_Float16*)(ws + WS_HBUF);
  float*    ypart = (float*)   (ws + WS_YPART);
  int*      bar   = (int*)     (ws + WS_BAR);

  k_init <<<64, 512, 0, stream>>>(H0, hbuf, bar);
  k_embed<<<SEQ * BATCH, EMB, 0, stream>>>(inputs, emb, x16);
  k_pack <<<4 * 32 * KT_TOTAL, 512, 0, stream>>>(W_xi, W_hi, W_xf, W_hf,
                                                 W_xo, W_ho, W_xc, W_hc, wpack);
  k_u    <<<1, HID, 0, stream>>>(W_hq, b_q, dw, db, u);
  k_lstm <<<NWG, 256, 0, stream>>>(x16, wpack, u, hbuf, ypart, bar, C0,
                                   b_i, b_f, b_o, b_c, out);
  k_reduce<<<(SEQ * BATCH) / 256, 256, 0, stream>>>(ypart, u, out);
}
